// Attention_18631568130798
// MI455X (gfx1250) — compile-verified, matrix operand order checked by reading, not run on hardware
//
#include <hip/hip_runtime.h>
#include <hip/hip_bf16.h>

typedef __attribute__((ext_vector_type(16))) _Float16 v16h;
typedef __attribute__((ext_vector_type(8)))  _Float16 v8h;
typedef __attribute__((ext_vector_type(2)))  _Float16 v2h;
typedef __attribute__((ext_vector_type(8)))  float    v8f;

#define Bt   64
#define Ntok 245
#define NT   49
#define Hh   12
#define HD   64
#define Cc   768
#define SCALE 0.125f

// Plain row-major layouts (halves):
//  qf,kf : [bh][256 tok][64 d]   (stride 64)  -> A frags for q, B frags for k^T
//  vT    : [bh][64 d][256 tok]   (stride 256) -> B frags for P@V
//  tvec  : [3136 row][768]       (stride 768)
//  aoutF : [15680 row][768]      (stride 768)

// Fragment of a row-major tile: lane(m=l&15,grp=l>>4) = two contiguous 8-half runs
// at [m][grp*8] and [m][16+grp*8]  (per CDNA5 ISA 16-bit A/B layout).
__device__ __forceinline__ v16h ldfrag(const _Float16* P, int stride, int lane) {
  const int m = lane & 15, grp = lane >> 4;
  const _Float16* p = P + m * stride + grp * 8;
  v8h lo = *(const v8h*)p;
  v8h hi = *(const v8h*)(p + 16);
  return __builtin_shufflevector(lo, hi, 0, 1, 2, 3, 4, 5, 6, 7, 8, 9, 10, 11, 12, 13, 14, 15);
}
__device__ __forceinline__ v8f wmma16(v16h a, v16h b, v8f c) {
  return __builtin_amdgcn_wmma_f32_16x16x32_f16(false, a, false, b, (short)0, c, false, false);
}
__device__ __forceinline__ v8h pk8(const float* f) {   // 4x v_cvt_pk_rtz_f16_f32
  v2h p0 = __builtin_bit_cast(v2h, __builtin_amdgcn_cvt_pkrtz(f[0], f[1]));
  v2h p1 = __builtin_bit_cast(v2h, __builtin_amdgcn_cvt_pkrtz(f[2], f[3]));
  v2h p2 = __builtin_bit_cast(v2h, __builtin_amdgcn_cvt_pkrtz(f[4], f[5]));
  v2h p3 = __builtin_bit_cast(v2h, __builtin_amdgcn_cvt_pkrtz(f[6], f[7]));
  v8h r;
  r[0] = p0[0]; r[1] = p0[1]; r[2] = p1[0]; r[3] = p1[1];
  r[4] = p2[0]; r[5] = p2[1]; r[6] = p3[0]; r[7] = p3[1];
  return r;
}

// =====================================================================
// Kernel 0: expand RPE bias -> biasx (H,245,256) f32 (shared by all 64 b)
// =====================================================================
__global__ __launch_bounds__(256) void rpe_expand(
    const float* __restrict__ rpe_table, const int* __restrict__ rpe_index,
    float* __restrict__ biasx, int rpeT)
{
  const int i = blockIdx.x, h = blockIdx.y, j = threadIdx.x;
  biasx[((long)h * Ntok + i) * 256 + j] =
      (j < Ntok) ? rpe_table[h * rpeT + rpe_index[i * Ntok + j]] : 0.f;
}

// =====================================================================
// Kernel 1: qkv GEMM -> q(*SCALE),k token-major / v d-major f16
// grid (245, 18), block 256. Tile 64(M) x 128(N), reg double-buffered.
// =====================================================================
__global__ __launch_bounds__(256) void qkv_gemm(
    const float* __restrict__ x, const float* __restrict__ w,
    const float* __restrict__ bias,
    _Float16* __restrict__ qf, _Float16* __restrict__ kf, _Float16* __restrict__ vT)
{
  __shared__ __align__(16) _Float16 lA[64 * 32];
  __shared__ __align__(16) _Float16 lB[128 * 32];
  const int t = threadIdx.x, lane = t & 31, wave = t >> 5;
  const int m0 = blockIdx.x * 64, n0 = blockIdx.y * 128;
  const int wm = wave & 1, wn = wave >> 1;
  const int arow = t >> 2, acol = (t & 3) * 8;     // A: 64x32, 8 halves/thread
  const int brow = t >> 1, bcol = (t & 1) * 16;    // B: 128x32, 16 halves/thread
  float regA[8], regB[16];
  auto loadAB = [&](int k0) {
    const float4* pa = (const float4*)&x[(m0 + arow) * Cc + k0 + acol];
    float4 a0 = pa[0], a1 = pa[1];
    regA[0] = a0.x; regA[1] = a0.y; regA[2] = a0.z; regA[3] = a0.w;
    regA[4] = a1.x; regA[5] = a1.y; regA[6] = a1.z; regA[7] = a1.w;
    const float4* pb = (const float4*)&w[(n0 + brow) * Cc + k0 + bcol];
    float4 b0 = pb[0], b1 = pb[1], b2 = pb[2], b3 = pb[3];
    regB[0] = b0.x; regB[1] = b0.y; regB[2]  = b0.z; regB[3]  = b0.w;
    regB[4] = b1.x; regB[5] = b1.y; regB[6]  = b1.z; regB[7]  = b1.w;
    regB[8] = b2.x; regB[9] = b2.y; regB[10] = b2.z; regB[11] = b2.w;
    regB[12] = b3.x; regB[13] = b3.y; regB[14] = b3.z; regB[15] = b3.w;
  };
  loadAB(0);
  v8f acc[2][2] = {};
  for (int k0 = 0; k0 < Cc; k0 += 32) {
    __syncthreads();
    *(v8h*)&lA[arow * 32 + acol]     = pk8(regA);
    *(v8h*)&lB[brow * 32 + bcol]     = pk8(regB);
    *(v8h*)&lB[brow * 32 + bcol + 8] = pk8(regB + 8);
    __syncthreads();
    if (k0 + 32 < Cc) loadAB(k0 + 32);           // overlap next tile with WMMAs
    v16h a[2], b[2];
    for (int i = 0; i < 2; ++i) a[i] = ldfrag(lA + (wm * 32 + i * 16) * 32, 32, lane);
    for (int j = 0; j < 2; ++j) b[j] = ldfrag(lB + (wn * 32 + j * 16) * 32, 32, lane);
    for (int i = 0; i < 2; ++i)
      for (int j = 0; j < 2; ++j)
        acc[i][j] = wmma16(a[i], b[j], acc[i][j]);
  }
  // Epilogue: row index is affine in r -> divide once per block, wrap per element.
  const int mrow = (lane >> 4) << 3, ncol = lane & 15;
  const int bidx0 = m0 / Ntok, tok0 = m0 % Ntok;   // block-uniform (SALU)
  for (int i = 0; i < 2; ++i)
    for (int j = 0; j < 2; ++j) {
      const int n = n0 + wn * 32 + j * 16 + ncol;
      const int which = n / Cc, c = n % Cc, h = c >> 6, d = c & 63;
      const float bv = bias[n];
      for (int r = 0; r < 8; ++r) {
        int m = wm * 32 + i * 16 + r + mrow;
        int tok = tok0 + m, bidx = bidx0;
        if (tok >= Ntok) { tok -= Ntok; ++bidx; }  // m<64 => at most one wrap
        float val = acc[i][j][r] + bv;
        long base = (long)(bidx * Hh + h) * 16384;
        if (which == 2)      vT[base + d * 256 + tok] = (_Float16)val;
        else if (which == 0) qf[base + tok * 64 + d] = (_Float16)(val * SCALE);
        else                 kf[base + tok * 64 + d] = (_Float16)val;
      }
    }
}

// =====================================================================
// Kernel 2: per (b,h) template attention -> t_vec row-major f16
// grid (768), block 128 (4 waves); wave w owns rows 16w..16w+15.
// =====================================================================
__global__ __launch_bounds__(128) void tmpl_attn(
    const _Float16* __restrict__ qf, const _Float16* __restrict__ kf,
    const float* __restrict__ boxmask, const float* __restrict__ biasx,
    _Float16* __restrict__ tvec)
{
  __shared__ float                  sc[4][16][64];
  __shared__ __align__(16) _Float16 pr[4][16][64];
  const int bh = blockIdx.x, b = bh / Hh, h = bh % Hh;
  const int t = threadIdx.x, lane = t & 31, wave = t >> 5;
  const int r0 = wave * 16;
  const _Float16* qbase = qf + (long)bh * 16384 + r0 * 64;
  v16h aq[2] = { ldfrag(qbase, 64, lane), ldfrag(qbase + 32, 64, lane) };
  v8f acc[4] = {};
  for (int ct = 0; ct < 4; ++ct) {
    const _Float16* kbase = kf + (long)bh * 16384 + ct * 16 * 64;
    for (int ks = 0; ks < 2; ++ks)
      acc[ct] = wmma16(aq[ks], ldfrag(kbase + ks * 32, 64, lane), acc[ct]);
  }
  const int mrow = (lane >> 4) << 3, ncol = lane & 15;
  for (int r = 0; r < 8; ++r) {
    const int i = r0 + r + mrow;
    const float* brow = biasx + (long)h * Ntok * 256 + (long)i * 256;
    for (int ct = 0; ct < 4; ++ct) {
      int jc = ct * 16 + ncol;
      float s = acc[ct][r];
      if (i < NT && jc < NT) s += brow[jc];
      sc[wave][r + mrow][ct * 16 + ncol] = s;
    }
  }
  __syncthreads();
  // masked softmax over full row: zeros outside 49x49 block contribute exp(0)=1
  if (t < 64) {
    const int m = t & 15, w2 = t >> 4;
    float* row = &sc[w2][m][0];
    _Float16* prow = &pr[w2][m][0];
    if (t < NT) {
      float mx = 0.f;
      for (int j = 0; j < NT; ++j) mx = fmaxf(mx, row[j]);
      float den = (float)(Ntok - NT) * __expf(-mx);
      for (int j = 0; j < NT; ++j) den += __expf(row[j] - mx);
      float inv = 1.f / den;
      for (int c8 = 0; c8 < 8; ++c8) {
        float tmp[8];
        for (int u = 0; u < 8; ++u) {
          int j = c8 * 8 + u;
          tmp[u] = (j < NT) ? __expf(row[j] - mx) * inv : 0.f;
        }
        *(v8h*)&prow[c8 * 8] = pk8(tmp);
      }
    } else {
      float z[8] = {};
      for (int c8 = 0; c8 < 8; ++c8) *(v8h*)&prow[c8 * 8] = pk8(z);
    }
  }
  __syncthreads();
  // out_t = P @ bm   (16x64, K=64)
  v16h ap[2] = { ldfrag(&pr[wave][0][0], 64, lane), ldfrag(&pr[wave][0][0] + 32, 64, lane) };
  v8f oc[4] = {};
  for (int dt = 0; dt < 4; ++dt)
    for (int ks = 0; ks < 2; ++ks) {
      v16h bb;
      for (int j = 0; j < 16; ++j) {               // small gather: 49x64 per (b,h), f32 src
        int grp = lane >> 4, n = lane & 15;
        int k2 = (j < 8) ? j : (j + 8);
        int ktok = ks * 32 + k2 + (grp << 3);
        bb[j] = (ktok < NT)
                    ? (_Float16)boxmask[(long)(b * NT + ktok) * Cc + h * HD + dt * 16 + n]
                    : (_Float16)0.f;
      }
      oc[dt] = wmma16(ap[ks], bb, oc[dt]);
    }
  for (int dt = 0; dt < 4; ++dt)
    for (int r = 0; r < 8; ++r) {
      int i = r0 + r + mrow;
      if (i < NT)
        tvec[(long)(b * NT + i) * Cc + h * HD + dt * 16 + ncol] = (_Float16)oc[dt][r];
    }
}

// =====================================================================
// Kernel 3: factor = t_vec @ box_w^T + box_b -> (B,H,49) f32
// grid (49), block 128 (4 waves, one 16x16 tile each). K=768.
// =====================================================================
__global__ __launch_bounds__(128) void box_factor(
    const _Float16* __restrict__ tvec, const float* __restrict__ box_w,
    const float* __restrict__ box_b, float* __restrict__ factor)
{
  __shared__ __align__(16) _Float16 lbw[16 * Cc];    // row-major 16x768
  const int t = threadIdx.x, lane = t & 31, wave = t >> 5;
  for (int i = 0; i < 12; ++i) {                     // 1536 chunks of 8 / 128 threads
    int c = t + i * 128, pos = c * 8, n = pos / Cc, k = pos % Cc;
    float tmp[8];
    for (int u = 0; u < 8; ++u) tmp[u] = (n < Hh) ? box_w[n * Cc + k + u] : 0.f;
    *(v8h*)&lbw[pos] = pk8(tmp);
  }
  __syncthreads();
  const int mt = blockIdx.x * 4 + wave;              // 196 tiles cover 3136 rows
  const _Float16* ablk = tvec + (long)mt * 16 * Cc;
  v8f acc = {};
  for (int ks = 0; ks < 24; ++ks)
    acc = wmma16(ldfrag(ablk + ks * 32, Cc, lane), ldfrag(lbw + ks * 32, Cc, lane), acc);
  const int mrow = (lane >> 4) << 3, hcol = lane & 15;
  if (hcol < Hh) {
    const int base = mt * 16;                        // divide once, wrap per element
    const int b0 = base / NT, iBase = base % NT;
    for (int r = 0; r < 8; ++r) {
      int i = iBase + r + mrow, b = b0;
      if (i >= NT) { i -= NT; ++b; }                 // span 16 < 49 => one wrap max
      factor[(b * Hh + hcol) * NT + i] = acc[r] + box_b[hcol];
    }
  }
}

// =====================================================================
// Kernel 4: full attention -> aout row-major f16
// grid (16, 12, 64): (rowTile, h, b). block 256 (8 waves).
// =====================================================================
__global__ __launch_bounds__(256) void attn_out(
    const _Float16* __restrict__ qf, const _Float16* __restrict__ kf,
    const _Float16* __restrict__ vT, const float* __restrict__ biasx,
    const float* __restrict__ factor, _Float16* __restrict__ aoutF)
{
  __shared__ float                  sc[16][256];
  __shared__ __align__(16) _Float16 pr[16][256];
  __shared__ float                  red[16][16];
  const int rt = blockIdx.x, h = blockIdx.y, b = blockIdx.z;
  const int bh = b * Hh + h, i0 = rt * 16;
  const int t = threadIdx.x, lane = t & 31, wave = t >> 5;
  const _Float16* qbase = qf + (long)bh * 16384 + i0 * 64;
  v16h aq[2] = { ldfrag(qbase, 64, lane), ldfrag(qbase + 32, 64, lane) };
  const int mrow = (lane >> 4) << 3, ncol = lane & 15;
  float fac_r[8];
  for (int r = 0; r < 8; ++r) {
    int i = i0 + r + mrow;
    fac_r[r] = (i < NT) ? factor[bh * NT + i] : 1.f;
  }
  v8f acc[2];
  for (int cc = 0; cc < 2; ++cc) {
    const int ct = wave * 2 + cc;                  // 16 col-tiles cover 256 cols
    const _Float16* kbase = kf + (long)bh * 16384 + ct * 16 * 64;
    acc[cc] = v8f{};
    for (int ks = 0; ks < 2; ++ks)
      acc[cc] = wmma16(aq[ks], ldfrag(kbase + ks * 32, 64, lane), acc[cc]);
  }
  for (int r = 0; r < 8; ++r) {
    const int i = i0 + r + mrow;
    const float* brow = biasx + (long)h * Ntok * 256 + (long)i * 256;
    for (int cc = 0; cc < 2; ++cc) {
      int jc = (wave * 2 + cc) * 16 + ncol;
      float s;
      if (i < Ntok && jc < Ntok) {
        s = acc[cc][r] + brow[jc];
        if (i < NT && jc >= NT) s *= fac_r[r];
      } else {
        s = -3.0e38f;
      }
      sc[r + mrow][(wave * 2 + cc) * 16 + ncol] = s;
    }
  }
  __syncthreads();
  // row softmax: 16 rows x 16 segments of 16 cols
  const int row = t >> 4, seg = t & 15;
  float mx = -3.0e38f;
  for (int j = 0; j < 16; ++j) mx = fmaxf(mx, sc[row][seg * 16 + j]);
  red[row][seg] = mx;
  __syncthreads();
  mx = red[row][0];
  for (int s2 = 1; s2 < 16; ++s2) mx = fmaxf(mx, red[row][s2]);
  float sum = 0.f;
  for (int j = 0; j < 16; ++j) sum += __expf(sc[row][seg * 16 + j] - mx);
  __syncthreads();
  red[row][seg] = sum;
  __syncthreads();
  sum = 0.f;
  for (int s2 = 0; s2 < 16; ++s2) sum += red[row][s2];
  float inv = 1.f / sum;
  for (int half = 0; half < 2; ++half) {
    float tmp[8];
    for (int u = 0; u < 8; ++u)
      tmp[u] = __expf(sc[row][seg * 16 + half * 8 + u] - mx) * inv;
    *(v8h*)&pr[row][seg * 16 + half * 8] = pk8(tmp);
  }
  __syncthreads();
  // out = P @ V  (16 x 64, K=256): waves 0..3 each own one 16-wide d tile
  if (wave < 4) {
    const _Float16* vbase = vT + (long)bh * 16384 + (wave * 16) * 256;
    v8f oc = {};
    for (int ks = 0; ks < 8; ++ks)
      oc = wmma16(ldfrag(&pr[0][0] + ks * 32, 256, lane),
                  ldfrag(vbase + ks * 32, 256, lane), oc);
    for (int r = 0; r < 8; ++r) {
      int i = i0 + r + mrow;
      if (i < Ntok)
        aoutF[(long)(b * Ntok + i) * Cc + h * HD + wave * 16 + ncol] = (_Float16)oc[r];
    }
  }
}

// =====================================================================
// Kernel 5: out = aout @ proj_w^T + proj_b (f32). A row-major f16 in
// global (no LDS staging); B staged via LDS, reg double-buffered.
// grid (245, 6), block 256.
// =====================================================================
__global__ __launch_bounds__(256) void proj_gemm(
    const _Float16* __restrict__ aoutF, const float* __restrict__ w,
    const float* __restrict__ bias, float* __restrict__ out)
{
  __shared__ __align__(16) _Float16 lB[128 * 32];
  const int t = threadIdx.x, lane = t & 31, wave = t >> 5;
  const int m0 = blockIdx.x * 64, n0 = blockIdx.y * 128;
  const int wm = wave & 1, wn = wave >> 1;
  const int brow = t >> 1, bcol = (t & 1) * 16;
  float regB[16];
  auto loadB = [&](int k0) {
    const float4* pb = (const float4*)&w[(n0 + brow) * Cc + k0 + bcol];
    float4 b0 = pb[0], b1 = pb[1], b2 = pb[2], b3 = pb[3];
    regB[0] = b0.x; regB[1] = b0.y; regB[2]  = b0.z; regB[3]  = b0.w;
    regB[4] = b1.x; regB[5] = b1.y; regB[6]  = b1.z; regB[7]  = b1.w;
    regB[8] = b2.x; regB[9] = b2.y; regB[10] = b2.z; regB[11] = b2.w;
    regB[12] = b3.x; regB[13] = b3.y; regB[14] = b3.z; regB[15] = b3.w;
  };
  loadB(0);
  v8f acc[2][2] = {};
  for (int k0 = 0; k0 < Cc; k0 += 32) {
    __syncthreads();
    *(v8h*)&lB[brow * 32 + bcol]     = pk8(regB);
    *(v8h*)&lB[brow * 32 + bcol + 8] = pk8(regB + 8);
    __syncthreads();
    if (k0 + 32 < Cc) loadB(k0 + 32);
    v16h a[2], b[2];
    for (int i = 0; i < 2; ++i)
      a[i] = ldfrag(aoutF + (long)(m0 + wm * 32 + i * 16) * Cc + k0, Cc, lane);
    for (int j = 0; j < 2; ++j) b[j] = ldfrag(lB + (wn * 32 + j * 16) * 32, 32, lane);
    for (int i = 0; i < 2; ++i)
      for (int j = 0; j < 2; ++j)
        acc[i][j] = wmma16(a[i], b[j], acc[i][j]);
  }
  const int mrow = (lane >> 4) << 3, ncol = lane & 15;
  for (int i = 0; i < 2; ++i)
    for (int j = 0; j < 2; ++j)
      for (int r = 0; r < 8; ++r) {
        int m = m0 + wm * 32 + i * 16 + r + mrow;
        int n = n0 + wn * 32 + j * 16 + ncol;
        out[(long)m * Cc + n] = acc[i][j][r] + bias[n];
      }
}

// =====================================================================
extern "C" void kernel_launch(void* const* d_in, const int* in_sizes, int n_in,
                              void* d_out, int out_size, void* d_ws, size_t ws_size,
                              hipStream_t stream) {
  const float* x        = (const float*)d_in[0];
  const float* boxmask  = (const float*)d_in[1];
  const float* qkv_w    = (const float*)d_in[2];
  const float* qkv_b    = (const float*)d_in[3];
  const float* proj_w   = (const float*)d_in[4];
  const float* proj_b   = (const float*)d_in[5];
  const float* box_w    = (const float*)d_in[6];
  const float* box_b    = (const float*)d_in[7];
  const float* rpe_tab  = (const float*)d_in[8];
  const int*   rpe_idx  = (const int*)d_in[9];
  const int    rpeT     = in_sizes[8] / Hh;
  float* out = (float*)d_out;

  const size_t nQKV = (size_t)Bt * Hh * 16384;          // q/k/vT halves each
  const size_t nTV  = (size_t)Bt * NT * Cc;             // tvec halves
  const size_t nAO  = (size_t)Bt * Ntok * Cc;           // aout halves
  char* ws = (char*)d_ws;
  _Float16* qf    = (_Float16*)ws;  ws += nQKV * 2;
  _Float16* kf    = (_Float16*)ws;  ws += nQKV * 2;
  _Float16* vT    = (_Float16*)ws;  ws += nQKV * 2;
  _Float16* tvec  = (_Float16*)ws;  ws += nTV * 2;
  _Float16* aoutF = (_Float16*)ws;  ws += nAO * 2;
  float*    biasx = (float*)ws;     ws += (size_t)Hh * Ntok * 256 * 4;
  float*    fac   = (float*)ws;     ws += (size_t)Bt * Hh * NT * 4;

  // zero padded token slots in q/k/v buffers (keeps pads finite/clean)
  (void)hipMemsetAsync(qf, 0, nQKV * 2, stream);
  (void)hipMemsetAsync(kf, 0, nQKV * 2, stream);
  (void)hipMemsetAsync(vT, 0, nQKV * 2, stream);

  rpe_expand<<<dim3(Ntok, Hh), 256, 0, stream>>>(rpe_tab, rpe_idx, biasx, rpeT);
  qkv_gemm<<<dim3(Ntok, 18), 256, 0, stream>>>(x, qkv_w, qkv_b, qf, kf, vT);
  tmpl_attn<<<dim3(Bt * Hh), 128, 0, stream>>>(qf, kf, boxmask, biasx, tvec);
  box_factor<<<dim3(NT), 128, 0, stream>>>(tvec, box_w, box_b, fac);
  attn_out<<<dim3(16, Hh, Bt), 256, 0, stream>>>(qf, kf, vT, biasx, fac, aoutF);
  proj_gemm<<<dim3(Ntok, 6), 256, 0, stream>>>(aoutF, proj_w, proj_b, out);
}
